// SGCN_25417616457795
// MI455X (gfx1250) — matrix operand order repaired and checked by reading, not compiled, not run on hardware
//
#include <hip/hip_runtime.h>

typedef __attribute__((ext_vector_type(16))) _Float16 v16h;
typedef __attribute__((ext_vector_type(8)))  _Float16 v8h;
typedef __attribute__((ext_vector_type(2)))  _Float16 v2h;
typedef __attribute__((ext_vector_type(8)))  float    v8f;

#define N_    64
#define CIN   64
#define T_    300
#define V_    25
#define KADJ  3
#define COUT  64
#define TT    16          // t-tile per workgroup
#define NTB   19          // ceil(300/16)

// LDS strides in f16 elements (padded for bank-conflict-free operand access)
#define ZSTR  200         // z[p][ki],  p = tt*32 + w  (512 rows)
#define XSTR  40          // X[m][v],   m = i*16 + tt  (1024 rows)
#define WSTR  200         // Wt[c][ki]  (64 rows)
#define ASTR  40          // At[k][w][v]

#define Z_ELEMS (512 * ZSTR)
#define X_ELEMS (1024 * XSTR)
#define W_ELEMS (64 * WSTR)
#define A_ELEMS (KADJ * 32 * ASTR)
#define SMEM_BYTES ((Z_ELEMS + X_ELEMS + W_ELEMS + A_ELEMS) * 2 + (KADJ * 32 + 192) * 4)

__device__ __forceinline__ v16h cat8(v8h lo, v8h hi) {
    return __builtin_shufflevector(lo, hi, 0,1,2,3,4,5,6,7,8,9,10,11,12,13,14,15);
}
__device__ __forceinline__ v16h frag32(const _Float16* p) {
    // two aligned 16B reads: k 0..7 at p, k 16..23 at p+16
    return cat8(*(const v8h*)p, *(const v8h*)(p + 16));
}
__device__ __forceinline__ v8f wmma16(v16h a, v16h b, v8f c) {
    return __builtin_amdgcn_wmma_f32_16x16x32_f16(false, a, false, b,
                                                  (short)0, c, false, false);
}

__global__ __launch_bounds__(256)
void sgcn_fused(const float* __restrict__ x, const float* __restrict__ A,
                const float* __restrict__ W, const float* __restrict__ b,
                float* __restrict__ out)
{
    extern __shared__ char smem_raw[];
    _Float16* Zl = (_Float16*)smem_raw;            // [512][ZSTR]
    _Float16* Xl = Zl + Z_ELEMS;                   // [1024][XSTR]
    _Float16* Wl = Xl + X_ELEMS;                   // [64][WSTR]
    _Float16* Al = Wl + W_ELEMS;                   // [KADJ][32][ASTR] (A^T: [w][v])
    float*    sA = (float*)(Al + A_ELEMS);         // [KADJ][32] column sums of A
    float*    bS = sA + KADJ * 32;                 // [192]

    const int n    = blockIdx.x;
    const int tb   = blockIdx.y;
    const int tid  = threadIdx.x;
    const int wv   = __builtin_amdgcn_readfirstlane(tid >> 5);  // scalar wave id
    const int lane = tid & 31;
    const int l15  = lane & 15;
    const int hi   = (lane >> 4) & 1;

    // ---------------- Phase 0: stage LDS (f32 -> f16 convert) ----------------
    // Wt[c][ki] = W[i][k*64 + c], ki = k*64 + i
    for (int idx = tid; idx < 64 * 192; idx += 256) {
        int c = idx / 192, ki = idx - c * 192;
        int k = ki >> 6, i = ki & 63;
        Wl[c * WSTR + ki] = (_Float16)W[i * 192 + k * 64 + c];
    }
    // At[k][w][v] = A[k][v][w], zero-padded to 32x32
    for (int idx = tid; idx < KADJ * 32 * ASTR; idx += 256) {
        int k = idx / (32 * ASTR); int r = idx - k * (32 * ASTR);
        int w = r / ASTR, v = r - w * ASTR;
        float val = (v < V_ && w < V_) ? A[k * (V_ * V_) + v * V_ + w] : 0.0f;
        Al[(k * 32 + w) * ASTR + v] = (_Float16)val;
    }
    if (tid < KADJ * 32) {
        int k = tid >> 5, w = tid & 31;
        float s = 0.0f;
        if (w < V_)
            for (int v = 0; v < V_; ++v) s += A[k * (V_ * V_) + v * V_ + w];
        sA[tid] = s;
    }
    if (tid < 192) bS[tid] = b[tid];
    // X[m][v] = x[n][i][tb*16+tt][v], m = i*16 + tt, zero-padded
    for (int idx = tid; idx < 1024 * 32; idx += 256) {
        int m = idx >> 5, v = idx & 31;
        int i = m >> 4, tt = m & 15;
        int t = tb * TT + tt;
        float val = (v < V_ && t < T_)
                  ? x[(((size_t)n * CIN + i) * T_ + t) * V_ + v] : 0.0f;
        Xl[m * XSTR + v] = (_Float16)val;
    }
    __syncthreads();

    // ---- Stage 1: z[(tt,w)][ki] = sum_v X[(i,tt)][v] * A[k][v][w] ----
    // 96 iterations total: (k x i-pair); each does 4 WMMAs sharing 2 A-frags
    // and 2 B-frags, then packed b32 stores (adjacent ki pair per dword).
    const int klA = hi ? 8 : 0;            // f16 A-operand per-lane K offset
    const int klB = hi ? 8 : 0;            // f16 B-operand per-lane K offset
    const int rb  = hi * 8;                // D-fragment row base
    for (int it = wv; it < KADJ * 32; it += 8) {
        int k  = it >> 5;                  // 0..2
        int ip = it & 31;                  // i-pair index
        int i0 = ip << 1;
        const _Float16* x0 = Xl + (i0 * 16 + l15) * XSTR + klA;
        const _Float16* x1 = x0 + 16 * XSTR;
        v16h a0 = cat8(*(const v8h*)x0, *(const v8h*)(x0 + 16));
        v16h a1 = cat8(*(const v8h*)x1, *(const v8h*)(x1 + 16));
        const _Float16* b0 = Al + (k * 32 + l15) * ASTR + klB;   // nt = 0 (w 0..15)
        const _Float16* b1 = b0 + 16 * ASTR;                     // nt = 1 (w 16..31)
        v16h bv0 = frag32(b0);
        v16h bv1 = frag32(b1);
        v8f c00 = {}, c01 = {}, c10 = {}, c11 = {};
        c00 = wmma16(a0, bv0, c00);
        c01 = wmma16(a0, bv1, c01);
        c10 = wmma16(a1, bv0, c10);
        c11 = wmma16(a1, bv1, c11);
        int ki0 = k * 64 + i0;             // even -> 4B-aligned packed store
        #pragma unroll
        for (int r = 0; r < 8; ++r) {
            int p0 = (r + rb) * 32 + l15;  // cols w = l15
            int p1 = p0 + 16;              // cols w = 16 + l15
            v2h pa; pa.x = (_Float16)c00[r]; pa.y = (_Float16)c10[r];
            v2h pb; pb.x = (_Float16)c01[r]; pb.y = (_Float16)c11[r];
            *(v2h*)(Zl + p0 * ZSTR + ki0) = pa;
            *(v2h*)(Zl + p1 * ZSTR + ki0) = pb;
        }
    }
    __syncthreads();

    // ---- Stage 2: out[c][(tt,w)] = sum_ki Wt[c][ki] * z[(tt,w)][ki] ----
    // Per c-tile: preload 6 A-fragments, reuse across the 4 n-tiles this
    // wave owns (K = 192 = 6 x 32 exact).
    for (int ct = 0; ct < 4; ++ct) {
        v16h af[6];
        const _Float16* wbase = Wl + (ct * 16 + l15) * WSTR + klA;
        #pragma unroll
        for (int ks = 0; ks < 6; ++ks) {
            const _Float16* wr = wbase + ks * 32;
            af[ks] = cat8(*(const v8h*)wr, *(const v8h*)(wr + 16));
        }
        for (int nt = wv; nt < 32; nt += 8) {
            int col = nt * 16 + l15;
            int w   = col & 31, tt = col >> 5;
            const _Float16* zbase = Zl + col * ZSTR + klB;
            v8f acc = {};
            #pragma unroll
            for (int ks = 0; ks < 6; ++ks)
                acc = wmma16(af[ks], frag32(zbase + ks * 32), acc);
            int  t     = tb * TT + tt;
            bool valid = (w < V_) && (t < T_);
            #pragma unroll
            for (int r = 0; r < 8; ++r) {
                int c = ct * 16 + r + rb;
                float bias = bS[c]       * sA[w]
                           + bS[64 + c]  * sA[32 + w]
                           + bS[128 + c] * sA[64 + w];
                if (valid)
                    out[(((size_t)n * COUT + c) * T_ + t) * V_ + w] = acc[r] + bias;
            }
        }
    }
}

extern "C" void kernel_launch(void* const* d_in, const int* in_sizes, int n_in,
                              void* d_out, int out_size, void* d_ws, size_t ws_size,
                              hipStream_t stream) {
    (void)in_sizes; (void)n_in; (void)out_size; (void)d_ws; (void)ws_size;
    const float* x = (const float*)d_in[0];
    const float* A = (const float*)d_in[1];
    const float* W = (const float*)d_in[2];
    const float* b = (const float*)d_in[3];
    float* out = (float*)d_out;

    hipFuncSetAttribute((const void*)sgcn_fused,
                        hipFuncAttributeMaxDynamicSharedMemorySize,
                        (int)SMEM_BYTES);
    dim3 grid(N_, NTB, 1);
    sgcn_fused<<<grid, 256, SMEM_BYTES, stream>>>(x, A, W, b, out);

    // tuple output tail: copy A verbatim after out
    hipMemcpyAsync(out + (size_t)N_ * COUT * T_ * V_, A,
                   (size_t)KADJ * V_ * V_ * sizeof(float),
                   hipMemcpyDeviceToDevice, stream);
}
